// MoELayer_11003706213000
// MI455X (gfx1250) — matrix operand order repaired and checked by reading, not compiled, not run on hardware
//
#include <hip/hip_runtime.h>
#include <hip/hip_bf16.h>

// ---------------- problem constants ----------------
constexpr int Bc = 4, Sc = 4096, Dc = 1024, Ec = 8, Hc = 2 * Dc;
constexpr int Tc = Bc * Sc;            // 16384 tokens
constexpr int HCc = 256;               // H streaming chunk
constexpr int MT = 64;                 // tokens per block (4 row tiles)
constexpr int NTHR = 512;              // 16 waves

typedef __attribute__((ext_vector_type(16))) __bf16 v16bf;
typedef __attribute__((ext_vector_type(8)))  float  v8f;

// ---------------- WMMA helpers ----------------
__device__ __forceinline__ v8f wmma_bf16(v16bf a, v16bf b, v8f c) {
  return __builtin_amdgcn_wmma_f32_16x16x32_bf16(
      false, a, false, b, (short)0, c, false, false);
}

// A-fragment (16x32 bf16): p0 -> K pairs {0..7}(lo)/{16..23}(hi),
// p1 -> K pairs {8..15}(lo)/{24..31}(hi); each 16B contiguous.
__device__ __forceinline__ v16bf load_a2(const __bf16* p0, const __bf16* p1) {
  union { v16bf v; unsigned int u[8]; } A;
#pragma unroll
  for (int j = 0; j < 4; ++j) A.u[j]     = *(const unsigned int*)(p0 + 2 * j);
#pragma unroll
  for (int j = 0; j < 4; ++j) A.u[4 + j] = *(const unsigned int*)(p1 + 2 * j);
  return A.v;
}

// LDS layout (both xs and hs):
//   physical row  pr = (logical_row & 15)*4 + (logical_row / 16)   [rt interleaved]
//   16B-chunk swizzle with key = logical_row & 15:
//   element index = pr*ROWSZ + 32*((col>>5) ^ (key>>2))*? ...  decomposed:
//   chunk = col>>3 ; chunk' = chunk ^ key ; addr = pr*ROWSZ + chunk'*8 + (col&7)
// For A-fragment loads (col = kk + hi*16 {+8}, kk % 32 == 0) this decomposes into
//   kt = ((kk>>5) ^ (m>>2)) << 5           (shared across rt and both halves)
//   lo0 = ((2*hi) ^ (m&3)) << 3 ; lo1 = ((2*hi+1) ^ (m&3)) << 3   (loop-invariant)

// ---------------- utility kernels ----------------
__global__ void zero_i32_kernel(int* p, int n) {
  int i = blockIdx.x * blockDim.x + threadIdx.x;
  if (i < n) p[i] = 0;
}

__global__ void zero_f32_kernel(float* __restrict__ p, long n4) {
  long i = (long)blockIdx.x * blockDim.x + threadIdx.x;
  if (i < n4) { float4 z = {0.f, 0.f, 0.f, 0.f}; ((float4*)p)[i] = z; }
}

// dst[e][c][r] = (bf16) src[e][r][c]
__global__ __launch_bounds__(256)
void transpose_cvt_kernel(const float* __restrict__ src, __bf16* __restrict__ dst,
                          int R, int C) {
  __shared__ float tile[32][33];
  int e = blockIdx.z;
  const float* s = src + (size_t)e * R * C;
  __bf16*      d = dst + (size_t)e * R * C;
  int c0 = blockIdx.x * 32, r0 = blockIdx.y * 32;
  int tx = threadIdx.x, ty = threadIdx.y;
#pragma unroll
  for (int i = 0; i < 4; ++i)
    tile[ty + i * 8][tx] = s[(size_t)(r0 + ty + i * 8) * C + c0 + tx];
  __syncthreads();
#pragma unroll
  for (int i = 0; i < 4; ++i)
    d[(size_t)(c0 + ty + i * 8) * R + r0 + tx] = (__bf16)tile[tx][ty + i * 8];
}

// ---------------- router: softmax + top-2 + gather lists ----------------
__global__ __launch_bounds__(256)
void router_kernel(const float* __restrict__ x, const float* __restrict__ rw,
                   const float* __restrict__ rb, int* __restrict__ counts,
                   int* __restrict__ tokl, float* __restrict__ gatel) {
  int lane = threadIdx.x & 31;
  int t = blockIdx.x * 8 + (threadIdx.x >> 5);   // one wave32 per token
  const float* xrow = x + (size_t)t * Dc;
  float acc[8] = {0.f, 0.f, 0.f, 0.f, 0.f, 0.f, 0.f, 0.f};
#pragma unroll 4
  for (int i = 0; i < Dc / 32; ++i) {
    int d = i * 32 + lane;
    float xv = xrow[d];
    float4 r0 = ((const float4*)(rw + d * 8))[0];
    float4 r1 = ((const float4*)(rw + d * 8))[1];
    acc[0] += xv * r0.x; acc[1] += xv * r0.y; acc[2] += xv * r0.z; acc[3] += xv * r0.w;
    acc[4] += xv * r1.x; acc[5] += xv * r1.y; acc[6] += xv * r1.z; acc[7] += xv * r1.w;
  }
#pragma unroll
  for (int e = 0; e < 8; ++e)
#pragma unroll
    for (int off = 16; off; off >>= 1) acc[e] += __shfl_xor(acc[e], off, 32);

  if (lane == 0) {
    float l[8], mx = -1e30f;
#pragma unroll
    for (int e = 0; e < 8; ++e) { l[e] = acc[e] + rb[e]; mx = fmaxf(mx, l[e]); }
    float p[8];
#pragma unroll
    for (int e = 0; e < 8; ++e) p[e] = __expf(l[e] - mx);
    int i1 = 0;
#pragma unroll
    for (int e = 1; e < 8; ++e) if (p[e] > p[i1]) i1 = e;
    int i2 = (i1 == 0) ? 1 : 0;
#pragma unroll
    for (int e = 0; e < 8; ++e) if (e != i1 && p[e] > p[i2]) i2 = e;
    float denom = p[i1] + p[i2];
    int s1 = atomicAdd(&counts[i1], 1);
    tokl[i1 * Tc + s1] = t; gatel[i1 * Tc + s1] = p[i1] / denom;
    int s2 = atomicAdd(&counts[i2], 1);
    tokl[i2 * Tc + s2] = t; gatel[i2 * Tc + s2] = p[i2] / denom;
  }
}

// ---------------- fused expert FFN (bf16 WMMA) ----------------
// block = 512 threads (16 waves), 64 tokens/block; grid (Tc/64, Ec)
// LDS: xs 64x1024 bf16 (128KB) + hs 64x256 bf16 (32KB) = exactly 160KB.
__global__ __launch_bounds__(NTHR)
void moe_ffn_kernel(const float* __restrict__ x,
                    const __bf16* __restrict__ w1T, const float* __restrict__ b1,
                    const __bf16* __restrict__ w2T, const float* __restrict__ b2,
                    const int* __restrict__ counts, const int* __restrict__ tokl,
                    const float* __restrict__ gatel, float* __restrict__ out) {
  __shared__ __align__(16) __bf16 xs[MT * Dc];    // 131072 B
  __shared__ __align__(16) __bf16 hs[MT * HCc];   //  32768 B

  int e = blockIdx.y;
  int cnt = counts[e];
  int row0 = blockIdx.x * MT;
  if (row0 >= cnt) return;                        // block-uniform exit
  int rows = min(MT, cnt - row0);

  int tid = threadIdx.x, wave = tid >> 5, lane = tid & 31;
  int m = lane & 15, hi = lane >> 4;
  int rxh = m >> 2, rxl = m & 3;
  int lo0 = ((2 * hi) ^ rxl) << 3;                // loop-invariant low parts
  int lo1 = ((2 * hi + 1) ^ rxl) << 3;

  // ---- stage x tile -> swizzled bf16 LDS, zero-fill invalid rows
  {
    int c4 = tid & 255;                           // float4 granule column
    int rbase = tid >> 8;                         // 0 or 1
#pragma unroll
    for (int j = 0; j < MT / 2; ++j) {
      int r = rbase + 2 * j;                      // logical row (token slot)
      float4 v = {0.f, 0.f, 0.f, 0.f};
      if (r < rows) {
        int tk = tokl[e * Tc + row0 + r];
        v = ((const float4*)(x + (size_t)tk * Dc))[c4];
      }
      union { __bf16 b[4]; uint2 u; } pk;
      pk.b[0] = (__bf16)v.x; pk.b[1] = (__bf16)v.y;
      pk.b[2] = (__bf16)v.z; pk.b[3] = (__bf16)v.w;
      int key = r & 15;
      int pr  = key * 4 + (r >> 4);               // interleaved physical row
      int idx = pr * Dc + ((((c4 >> 1) ^ key)) << 3) + (c4 & 1) * 4;
      *(uint2*)&xs[idx] = pk.u;
    }
  }
  __syncthreads();

  v8f c[16] = {};   // [rt(4)][ct(4)] : wave owns out cols [wave*64,+64), rows 0..63

  // invariant A-fragment base pointers (physical row m*4, rt via immediate)
  const __bf16* xa0 = xs + m * 4 * Dc + lo0;
  const __bf16* xa1 = xs + m * 4 * Dc + lo1;
  const __bf16* ha0 = hs + m * 4 * HCc + lo0;
  const __bf16* ha1 = hs + m * 4 * HCc + lo1;

  for (int hc = 0; hc < Hc; hc += HCc) {
    // ---- phase 1: hs = relu(xs @ w1[:, hc:+256] + b1); 1 col tile x 4 row tiles
    {
      int nl = wave * 16;
      const __bf16* brow = w1T + ((size_t)e * Hc + hc + nl + m) * Dc + hi * 16;
      v8f acc[4] = {};
      for (int kk = 0; kk < Dc; kk += 32) {
        int kt = (((kk >> 5) ^ rxh) << 5);        // shared swizzle term
        v16bf b = *(const v16bf*)(brow + kk);
        const __bf16* p0 = xa0 + kt;
        const __bf16* p1 = xa1 + kt;
#pragma unroll
        for (int rt = 0; rt < 4; ++rt) {          // rt*Dc = 2048B immediates
          v16bf a = load_a2(p0 + rt * Dc, p1 + rt * Dc);
          acc[rt] = wmma_bf16(a, b, acc[rt]);
        }
      }
      float bias = b1[e * Hc + hc + nl + m];
      int col = nl + m;
#pragma unroll
      for (int rt = 0; rt < 4; ++rt)
#pragma unroll
        for (int v = 0; v < 8; ++v) {
          int key = hi ? v + 8 : v;               // logical row & 15
          float h = acc[rt][v] + bias;
          h = h > 0.f ? h : 0.f;
          int sidx = (key * 4 + rt) * HCc + ((((col >> 3) ^ key)) << 3) + (col & 7);
          hs[sidx] = (__bf16)h;
        }
    }
    __syncthreads();

    // ---- phase 2: c += hs @ w2[hc:+256, wave cols]; K outer for A reuse
    {
      const __bf16* bbase = w2T + ((size_t)e * Dc + wave * 64 + m) * Hc + hc + hi * 16;
#pragma unroll
      for (int kk = 0; kk < HCc; kk += 32) {
        int kt = (((kk >> 5) ^ rxh) << 5);
        const __bf16* p0 = ha0 + kt;
        const __bf16* p1 = ha1 + kt;
        v16bf a[4];
#pragma unroll
        for (int rt = 0; rt < 4; ++rt)            // rt*HCc = 512B immediates
          a[rt] = load_a2(p0 + rt * HCc, p1 + rt * HCc);
#pragma unroll
        for (int ct = 0; ct < 4; ++ct) {
          v16bf b = *(const v16bf*)(bbase + (size_t)ct * 16 * Hc + kk);
#pragma unroll
          for (int rt = 0; rt < 4; ++rt)
            c[rt * 4 + ct] = wmma_bf16(a[rt], b, c[rt * 4 + ct]);
        }
      }
    }
    __syncthreads();
  }

  // ---- epilogue: out[tok[mr], n] += gate[mr] * (y + b2)
#pragma unroll
  for (int rt = 0; rt < 4; ++rt)
#pragma unroll
    for (int v = 0; v < 8; ++v) {
      int mr = rt * 16 + (hi ? v + 8 : v);
      if (mr < rows) {
        int   tk = tokl[e * Tc + row0 + mr];
        float g  = gatel[e * Tc + row0 + mr];
        float* orow = out + (size_t)tk * Dc;
#pragma unroll
        for (int ct = 0; ct < 4; ++ct) {
          int n = wave * 64 + ct * 16 + m;
          float val = (c[rt * 4 + ct][v] + b2[e * Dc + n]) * g;
          (void)__hip_atomic_fetch_add(&orow[n], val,
                                       __ATOMIC_RELAXED, __HIP_MEMORY_SCOPE_AGENT);
        }
      }
    }
}

// ---------------- host launch ----------------
extern "C" void kernel_launch(void* const* d_in, const int* in_sizes, int n_in,
                              void* d_out, int out_size, void* d_ws, size_t ws_size,
                              hipStream_t stream) {
  const float* x        = (const float*)d_in[0];
  const float* router_w = (const float*)d_in[1];
  const float* router_b = (const float*)d_in[2];
  const float* w1       = (const float*)d_in[3];
  const float* b1       = (const float*)d_in[4];
  const float* w2       = (const float*)d_in[5];
  const float* b2       = (const float*)d_in[6];
  float* out = (float*)d_out;

  char* ws = (char*)d_ws;
  int*    counts = (int*)ws;
  int*    tokl   = (int*)(ws + 256);
  float*  gatel  = (float*)(ws + 256 + (size_t)Ec * Tc * 4);
  size_t  wOff   = 256 + 2 * (size_t)Ec * Tc * 4;
  wOff = (wOff + 255) & ~(size_t)255;
  __bf16* w1T = (__bf16*)(ws + wOff);                                  // [E][H][D]
  __bf16* w2T = (__bf16*)(ws + wOff + (size_t)Ec * Hc * Dc * 2);       // [E][D][H]

  zero_i32_kernel<<<1, 32, 0, stream>>>(counts, Ec);
  long n4 = (long)Tc * Dc / 4;
  zero_f32_kernel<<<(int)((n4 + 255) / 256), 256, 0, stream>>>(out, n4);

  transpose_cvt_kernel<<<dim3(Hc / 32, Dc / 32, Ec), dim3(32, 8), 0, stream>>>(w1, w1T, Dc, Hc);
  transpose_cvt_kernel<<<dim3(Dc / 32, Hc / 32, Ec), dim3(32, 8), 0, stream>>>(w2, w2T, Hc, Dc);

  router_kernel<<<Tc / 8, 256, 0, stream>>>(x, router_w, router_b, counts, tokl, gatel);

  moe_ffn_kernel<<<dim3(Tc / MT, Ec), NTHR, 0, stream>>>(x, w1T, b1, w2T, b2,
                                                         counts, tokl, gatel, out);
}
